// MultiSpeciesHead_25769804210
// MI455X (gfx1250) — matrix operand order repaired and checked by reading, not compiled, not run on hardware
//
#include <hip/hip_runtime.h>
#include <hip/hip_bf16.h>

typedef __bf16 bf16_t;
typedef __attribute__((ext_vector_type(4)))  __bf16 v4bf;
typedef __attribute__((ext_vector_type(8)))  __bf16 v8bf;
typedef __attribute__((ext_vector_type(16))) __bf16 v16bf;
typedef __attribute__((ext_vector_type(8)))  float  v8f;

#define B_DIM   4
#define S_DIM   896
#define D_DIM   1536
#define NMAX    5313
#define N0      5313
#define N1      1643
#define N2      128
#define LN_EPS  1e-5f
#define LDA     40                   // LDS row stride (bf16): 80B rows, conflict-free
#define SLAB_ELEMS (128 * LDA)       // 5120 bf16 per tile array
#define SLAB_BYTES (SLAB_ELEMS * 2)  // 10240 B
#define BUF_ELEMS  (4 * SLAB_ELEMS)  // Ah, Al, Bh, Bl
#define BUF_BYTES  (BUF_ELEMS * 2)   // 40960 B
#define KSTEPS     (D_DIM / 32)      // 48

// ---------------------------------------------------------------------------
// CDNA5 async global->LDS copy (ASYNCcnt-tracked, bypasses VGPRs).
// vdst = VGPR with LDS byte address, vaddr = 64-bit global address (GV mode).
// ---------------------------------------------------------------------------
__device__ __forceinline__ void cp_async_b128(uint32_t lds_addr, uint64_t gaddr) {
    asm volatile("global_load_async_to_lds_b128 %0, %1, off"
                 :: "v"(lds_addr), "v"(gaddr) : "memory");
}
__device__ __forceinline__ void wait_asynccnt0() {
    asm volatile("s_wait_asynccnt 0" ::: "memory");
}

// ---------------------------------------------------------------------------
// Kernel 0: split fp32 weights into bf16 hi + lo (one-shot, L2-resident after).
// ---------------------------------------------------------------------------
__global__ __launch_bounds__(256) void w_split_kernel(
    const float* __restrict__ W, bf16_t* __restrict__ wh, bf16_t* __restrict__ wl,
    int nChunk4)
{
    const int i = blockIdx.x * 256 + threadIdx.x;
    if (i >= nChunk4) return;
    const float4 wv = ((const float4*)W)[i];
    const float f[4] = {wv.x, wv.y, wv.z, wv.w};
    v4bf hv, lv;
#pragma unroll
    for (int e = 0; e < 4; ++e) {
        const bf16_t h = (bf16_t)f[e];
        hv[e] = h;
        lv[e] = (bf16_t)(f[e] - (float)h);
    }
    ((v4bf*)wh)[i] = hv;
    ((v4bf*)wl)[i] = lv;
}

// ---------------------------------------------------------------------------
// Kernel 1: fp32 LayerNorm, split normalized value into bf16 hi + lo.
// ---------------------------------------------------------------------------
__global__ __launch_bounds__(256) void ln_split_kernel(
    const float* __restrict__ x, const int* __restrict__ head_idx,
    const float* __restrict__ g0, const float* __restrict__ be0,
    const float* __restrict__ g1, const float* __restrict__ be1,
    const float* __restrict__ g2, const float* __restrict__ be2,
    bf16_t* __restrict__ xh, bf16_t* __restrict__ xl)
{
    const int row = blockIdx.x;               // 0 .. B*S-1
    const int b   = row / S_DIM;
    int head = head_idx[b];
    if (head > 2) head = 0;                   // ZeroHead: values never read downstream
    const float* gamma = (head == 0) ? g0 : (head == 1) ? g1 : g2;
    const float* beta  = (head == 0) ? be0 : (head == 1) ? be1 : be2;

    const float* xr = x + (size_t)row * D_DIM;
    float v[6];
    float s = 0.f, ss = 0.f;
#pragma unroll
    for (int j = 0; j < 6; ++j) {
        v[j] = xr[threadIdx.x + j * 256];
        s  += v[j];
        ss += v[j] * v[j];
    }
#pragma unroll
    for (int o = 16; o > 0; o >>= 1) {        // wave32 reduction
        s  += __shfl_xor(s,  o, 32);
        ss += __shfl_xor(ss, o, 32);
    }
    __shared__ float reds[8], redss[8];
    const int wave = threadIdx.x >> 5;
    if ((threadIdx.x & 31) == 0) { reds[wave] = s; redss[wave] = ss; }
    __syncthreads();
    s = 0.f; ss = 0.f;
#pragma unroll
    for (int w = 0; w < 8; ++w) { s += reds[w]; ss += redss[w]; }

    const float mu   = s  * (1.f / D_DIM);
    const float var  = ss * (1.f / D_DIM) - mu * mu;   // biased variance
    const float rinv = rsqrtf(var + LN_EPS);

#pragma unroll
    for (int j = 0; j < 6; ++j) {
        const int d = threadIdx.x + j * 256;
        const float xn = (v[j] - mu) * rinv * gamma[d] + beta[d];
        const bf16_t hi = (bf16_t)xn;
        const bf16_t lo = (bf16_t)(xn - (float)hi);
        xh[(size_t)row * D_DIM + d] = hi;
        xl[(size_t)row * D_DIM + d] = lo;
    }
}

__device__ __forceinline__ float softplusf(float z) {
    return (z > 20.f) ? z : log1pf(expf(z));
}

// ---------------------------------------------------------------------------
// Kernel 2: bf16x3 WMMA GEMM, 128x128 tile, K slab 32, async double-buffered.
// ---------------------------------------------------------------------------
__global__ __launch_bounds__(256) void head_gemm_kernel(
    const bf16_t* __restrict__ xh, const bf16_t* __restrict__ xl,
    const bf16_t* __restrict__ Wh, const bf16_t* __restrict__ Wl,
    const int* __restrict__ head_idx,
    const float* __restrict__ bi0, const float* __restrict__ bi1,
    const float* __restrict__ bi2,
    float* __restrict__ out)
{
    const int b  = blockIdx.z;
    const int n0 = blockIdx.x * 128;          // output column tile base
    const int m0 = blockIdx.y * 128;          // row tile base within batch
    float* outTile = out + ((size_t)b * S_DIM + m0) * NMAX;

    int n = 0; size_t headOff = 0; const float* bias = nullptr;
    const int head = head_idx[b];
    if      (head == 0) { n = N0; headOff = 0;                          bias = bi0; }
    else if (head == 1) { n = N1; headOff = (size_t)N0 * D_DIM;         bias = bi1; }
    else if (head == 2) { n = N2; headOff = (size_t)(N0 + N1) * D_DIM;  bias = bi2; }
    // head == 3 -> n = 0 -> pure zero fill

    if (n0 >= n) {   // entire tile is padding / ZeroHead
        const int ncol = (NMAX - n0 < 128) ? (NMAX - n0) : 128;
        for (int r = 0; r < 128; ++r)
            for (int c = threadIdx.x; c < ncol; c += 256)
                outTile[(size_t)r * NMAX + n0 + c] = 0.f;
        return;
    }

    __shared__ __align__(16) bf16_t smem[2 * BUF_ELEMS];   // 81920 B

    const int tid   = threadIdx.x;
    const int lane  = tid & 31;
    const int wave  = tid >> 5;
    const int wm    = wave >> 2;     // 0..1  (M)
    const int wn    = wave & 3;      // 0..3  (N)
    const int lr    = lane & 15;
    const int lhalf = lane >> 4;

    // ---- staging geometry: 512 16B-chunks per array, 2 per thread ----
    const int r0  = tid >> 2;            // rows r0 and r0+64
    const int kq8 = (tid & 3) * 8;       // bf16 elem offset within 32-K slab
    const uint32_t smem32 = (uint32_t)(uintptr_t)&smem[0];     // LDS byte base
    const uint32_t ldB0 = (uint32_t)((r0 * LDA + kq8) * 2);
    const uint32_t ldB1 = (uint32_t)(((r0 + 64) * LDA + kq8) * 2);

    const size_t aRowBase = (size_t)b * S_DIM + m0;
    uint64_t gAh0 = (uint64_t)(uintptr_t)(xh + (aRowBase + r0)      * D_DIM + kq8);
    uint64_t gAh1 = (uint64_t)(uintptr_t)(xh + (aRowBase + r0 + 64) * D_DIM + kq8);
    uint64_t gAl0 = (uint64_t)(uintptr_t)(xl + (aRowBase + r0)      * D_DIM + kq8);
    uint64_t gAl1 = (uint64_t)(uintptr_t)(xl + (aRowBase + r0 + 64) * D_DIM + kq8);
    const size_t bRowBase = headOff + (size_t)(n0 + r0) * D_DIM + kq8;
    uint64_t gBh0 = (uint64_t)(uintptr_t)(Wh + bRowBase);
    uint64_t gBh1 = (uint64_t)(uintptr_t)(Wh + bRowBase + (size_t)64 * D_DIM);
    uint64_t gBl0 = (uint64_t)(uintptr_t)(Wl + bRowBase);
    uint64_t gBl1 = (uint64_t)(uintptr_t)(Wl + bRowBase + (size_t)64 * D_DIM);

    v8f acc[4][2];
    const v8f vzero = {0.f, 0.f, 0.f, 0.f, 0.f, 0.f, 0.f, 0.f};
#pragma unroll
    for (int mi = 0; mi < 4; ++mi)
#pragma unroll
        for (int ni = 0; ni < 2; ++ni) acc[mi][ni] = vzero;

    // issue slab 0 into buffer 0
    {
        const uint32_t base = smem32;
        cp_async_b128(base + 0 * SLAB_BYTES + ldB0, gAh0);
        cp_async_b128(base + 0 * SLAB_BYTES + ldB1, gAh1);
        cp_async_b128(base + 1 * SLAB_BYTES + ldB0, gAl0);
        cp_async_b128(base + 1 * SLAB_BYTES + ldB1, gAl1);
        cp_async_b128(base + 2 * SLAB_BYTES + ldB0, gBh0);
        cp_async_b128(base + 2 * SLAB_BYTES + ldB1, gBh1);
        cp_async_b128(base + 3 * SLAB_BYTES + ldB0, gBl0);
        cp_async_b128(base + 3 * SLAB_BYTES + ldB1, gBl1);
    }

    for (int ks = 0; ks < KSTEPS; ++ks) {
        wait_asynccnt0();        // this wave's slab ks is in LDS
        __syncthreads();         // everyone's slab ks is in; other buffer free

        if (ks + 1 < KSTEPS) {   // prefetch slab ks+1 into the other buffer
            gAh0 += 64; gAh1 += 64; gAl0 += 64; gAl1 += 64;
            gBh0 += 64; gBh1 += 64; gBl0 += 64; gBl1 += 64;
            const uint32_t base = smem32 + (uint32_t)(((ks + 1) & 1) * BUF_BYTES);
            cp_async_b128(base + 0 * SLAB_BYTES + ldB0, gAh0);
            cp_async_b128(base + 0 * SLAB_BYTES + ldB1, gAh1);
            cp_async_b128(base + 1 * SLAB_BYTES + ldB0, gAl0);
            cp_async_b128(base + 1 * SLAB_BYTES + ldB1, gAl1);
            cp_async_b128(base + 2 * SLAB_BYTES + ldB0, gBh0);
            cp_async_b128(base + 2 * SLAB_BYTES + ldB1, gBh1);
            cp_async_b128(base + 3 * SLAB_BYTES + ldB0, gBl0);
            cp_async_b128(base + 3 * SLAB_BYTES + ldB1, gBl1);
        }

        const bf16_t* sAh = &smem[(ks & 1) * BUF_ELEMS];
        const bf16_t* sAl = sAh + SLAB_ELEMS;
        const bf16_t* sBh = sAl + SLAB_ELEMS;
        const bf16_t* sBl = sBh + SLAB_ELEMS;

        // ---- B fragments (resident across M subtiles) ----
        v16bf bh[2], bl[2];
#pragma unroll
        for (int ni = 0; ni < 2; ++ni) {
            const int t   = wn * 32 + ni * 16 + lr;
            const int off = t * LDA + lhalf * 16;     // contiguous K16 per lane half
            v8bf p0 = *(const v8bf*)&sBh[off];
            v8bf p1 = *(const v8bf*)&sBh[off + 8];
            bh[ni] = __builtin_shufflevector(p0, p1, 0,1,2,3,4,5,6,7,8,9,10,11,12,13,14,15);
            p0 = *(const v8bf*)&sBl[off];
            p1 = *(const v8bf*)&sBl[off + 8];
            bl[ni] = __builtin_shufflevector(p0, p1, 0,1,2,3,4,5,6,7,8,9,10,11,12,13,14,15);
        }
        // ---- A fragments + bf16x3 WMMA ----
#pragma unroll
        for (int mi = 0; mi < 4; ++mi) {
            const int m  = wm * 64 + mi * 16 + lr;
            const int o1 = m * LDA + lhalf * 16;      // K {h*16 .. h*16+7}
            const int o2 = o1 + 8;                    // K {h*16+8 ..}
            v8bf a0 = *(const v8bf*)&sAh[o1];
            v8bf a1 = *(const v8bf*)&sAh[o2];
            v16bf ah = __builtin_shufflevector(a0, a1, 0,1,2,3,4,5,6,7,8,9,10,11,12,13,14,15);
            a0 = *(const v8bf*)&sAl[o1];
            a1 = *(const v8bf*)&sAl[o2];
            v16bf al = __builtin_shufflevector(a0, a1, 0,1,2,3,4,5,6,7,8,9,10,11,12,13,14,15);
#pragma unroll
            for (int ni = 0; ni < 2; ++ni) {
                acc[mi][ni] = __builtin_amdgcn_wmma_f32_16x16x32_bf16(
                    false, ah, false, bh[ni], (short)0, acc[mi][ni], false, false);
                acc[mi][ni] = __builtin_amdgcn_wmma_f32_16x16x32_bf16(
                    false, ah, false, bl[ni], (short)0, acc[mi][ni], false, false);
                acc[mi][ni] = __builtin_amdgcn_wmma_f32_16x16x32_bf16(
                    false, al, false, bh[ni], (short)0, acc[mi][ni], false, false);
            }
        }
    }

    // ---- epilogue: bias + softplus, zero padding, clip at NMAX ----
#pragma unroll
    for (int ni = 0; ni < 2; ++ni) {
        const int tcol = n0 + wn * 32 + ni * 16 + lr;
        if (tcol >= NMAX) continue;
        const bool valid = (tcol < n);
        const float bia = valid ? bias[tcol] : 0.f;
#pragma unroll
        for (int mi = 0; mi < 4; ++mi) {
#pragma unroll
            for (int r = 0; r < 8; ++r) {
                const int m = wm * 64 + mi * 16 + lhalf * 8 + r;  // C/D layout
                float vv = 0.f;
                if (valid) vv = softplusf(acc[mi][ni][r] + bia);
                outTile[(size_t)m * NMAX + tcol] = vv;
            }
        }
    }
}

// ---------------------------------------------------------------------------
extern "C" void kernel_launch(void* const* d_in, const int* in_sizes, int n_in,
                              void* d_out, int out_size, void* d_ws, size_t ws_size,
                              hipStream_t stream) {
    const float* x        = (const float*)d_in[0];
    const int*   head_idx = (const int*)  d_in[1];
    const float* g0  = (const float*)d_in[2],  *be0 = (const float*)d_in[3];
    const float* W0  = (const float*)d_in[4],  *bi0 = (const float*)d_in[5];
    const float* g1  = (const float*)d_in[6],  *be1 = (const float*)d_in[7];
    const float* W1  = (const float*)d_in[8],  *bi1 = (const float*)d_in[9];
    const float* g2  = (const float*)d_in[10], *be2 = (const float*)d_in[11];
    const float* W2  = (const float*)d_in[12], *bi2 = (const float*)d_in[13];
    float* out = (float*)d_out;

    // ws layout (bf16): Wh[WTOT] | Wl[WTOT] | xh[XTOT] | xl[XTOT]  (~66 MB)
    const size_t WTOT = (size_t)(N0 + N1 + N2) * D_DIM;
    const size_t XTOT = (size_t)B_DIM * S_DIM * D_DIM;
    bf16_t* Wh = (bf16_t*)d_ws;
    bf16_t* Wl = Wh + WTOT;
    bf16_t* xh = Wl + WTOT;
    bf16_t* xl = xh + XTOT;

    const int c0 = N0 * D_DIM / 4, c1 = N1 * D_DIM / 4, c2 = N2 * D_DIM / 4;
    w_split_kernel<<<(c0 + 255) / 256, 256, 0, stream>>>(W0, Wh, Wl, c0);
    w_split_kernel<<<(c1 + 255) / 256, 256, 0, stream>>>(
        W1, Wh + (size_t)N0 * D_DIM, Wl + (size_t)N0 * D_DIM, c1);
    w_split_kernel<<<(c2 + 255) / 256, 256, 0, stream>>>(
        W2, Wh + (size_t)(N0 + N1) * D_DIM, Wl + (size_t)(N0 + N1) * D_DIM, c2);

    ln_split_kernel<<<B_DIM * S_DIM, 256, 0, stream>>>(
        x, head_idx, g0, be0, g1, be1, g2, be2, xh, xl);

    dim3 grid((NMAX + 127) / 128, S_DIM / 128, B_DIM);   // 42 x 7 x 4
    head_gemm_kernel<<<grid, 256, 0, stream>>>(
        xh, xl, Wh, Wl, head_idx, bi0, bi1, bi2, out);
}
